// GATModule_87917980549657
// MI455X (gfx1250) — matrix operand order repaired and checked by reading, not compiled, not run on hardware
//
#include <hip/hip_runtime.h>
#include <hip/hip_bf16.h>
#include <stdint.h>
#include <stddef.h>

// ---------------------------------------------------------------------------
// Types for CDNA5 WMMA (wave32): bf16 A/B (16 halves = 8 VGPRs), f32 C/D (8 VGPRs)
// ---------------------------------------------------------------------------
typedef __bf16 bf16_t;
typedef __attribute__((ext_vector_type(16))) __bf16 v16bf;
typedef __attribute__((ext_vector_type(8)))  __bf16 v8bf;
typedef __attribute__((ext_vector_type(8)))  float  v8f;

#define NEG_SLOPE 0.2f

__device__ __forceinline__ bf16_t f2bf(float f) {
  unsigned u = __float_as_uint(f);
  u += 0x7FFFu + ((u >> 16) & 1u);                 // round-to-nearest-even
  unsigned short s = (unsigned short)(u >> 16);
  return __builtin_bit_cast(bf16_t, s);
}

// order-preserving float -> uint key (for atomicMax-based segment max)
__device__ __forceinline__ unsigned fkey(float f) {
  int i = __float_as_int(f);
  unsigned u = (unsigned)i;
  return (i >= 0) ? (u | 0x80000000u) : ~u;
}
__device__ __forceinline__ float keyf(unsigned u) {
  unsigned b = (u & 0x80000000u) ? (u & 0x7FFFFFFFu) : ~u;
  return __uint_as_float(b);
}

// ---------------------------------------------------------------------------
// Weight packer: W[N,K] (row-major, = lin.weight [out,in]) -> Bp in WMMA
// B-fragment order [K/32][N/16][lane 0..31][half 0..15]:
//   lane l: n = nb*16 + (l&15); half h -> k = kb*32 + (l>>4)*16 + h
// so each lane's 16 halves are one contiguous 32-byte chunk.
// ---------------------------------------------------------------------------
__global__ void pack_b(const float* __restrict__ W, bf16_t* __restrict__ Bp,
                       int Nn, int K) {
  int idx = blockIdx.x * blockDim.x + threadIdx.x;
  if (idx >= Nn * K) return;
  int f = idx >> 9;                 // fragment index (512 halves per 32x16 tile)
  int r = idx & 511;
  int lane = r >> 4, half = r & 15;
  int tiles_n = Nn >> 4;
  int kb = f / tiles_n, nb = f - kb * tiles_n;
  int n = (nb << 4) + (lane & 15);
  int k = (kb << 5) + ((lane >> 4) << 4) + half;
  Bp[idx] = f2bf(W[(size_t)n * K + k]);
}

// ---------------------------------------------------------------------------
// bf16 WMMA GEMM, fragment-packed B, 16x64 strip per wave (4 accumulators,
// A fragment reused across 4 WMMAs).  D[M,N] = act(A[M,K] @ B + bias)
// Requires: M % 16 == 0, N % 64 == 0, K % 32 == 0 (true for every layer here).
// act: 0=none, 1=relu, 2=prelu (slope read from *pa on device)
// ---------------------------------------------------------------------------
__global__ __launch_bounds__(256)
void wmma_gemm_bf16(const bf16_t* __restrict__ A, const bf16_t* __restrict__ Bp,
                    const float* __restrict__ bias, const float* __restrict__ pa,
                    float* __restrict__ D, bf16_t* __restrict__ Dbf,
                    int M, int N, int K, int act) {
  const int wave = threadIdx.x >> 5;
  const int lane = threadIdx.x & 31;
  const int groups_n = N >> 6;                   // 64-wide N strips
  const int tile = blockIdx.x * 8 + wave;
  if (tile >= (M >> 4) * groups_n) return;       // wave-uniform guard (EXEC stays full)

  const int tm = (tile / groups_n) << 4;
  const int tg = tile % groups_n;
  const int tn = tg << 6;
  const int hl = lane >> 4;                      // half-wave select
  const int l  = lane & 15;

  const bf16_t* __restrict__ Arow = A + (size_t)(tm + l) * K + hl * 16;
  const bf16_t* __restrict__ bbase =
      Bp + (size_t)(tg * 4) * 512 + (size_t)lane * 16;
  const size_t bstep = (size_t)(N >> 4) * 512;   // advance one K-block of fragments

  v8f acc[4] = {};
  for (int k0 = 0; k0 < K; k0 += 32) {
    const bf16_t* ap = Arow + k0;
    __builtin_prefetch(ap + 64, 0, 1);           // global_prefetch_b8 next A block
    v8bf a0 = *(const v8bf*)(ap);
    v8bf a1 = *(const v8bf*)(ap + 8);
    v16bf a = __builtin_shufflevector(a0, a1, 0, 1, 2, 3, 4, 5, 6, 7,
                                      8, 9, 10, 11, 12, 13, 14, 15);
    const bf16_t* bp = bbase;
    bbase += bstep;
#pragma unroll
    for (int g = 0; g < 4; ++g) {
      v16bf b = *(const v16bf*)(bp + (size_t)g * 512);
      acc[g] = __builtin_amdgcn_wmma_f32_16x16x32_bf16(false, a, false, b,
                                                       (short)0, acc[g], false, false);
    }
  }

  const int m0 = tm + hl * 8;
  const float slope = (act == 2) ? pa[0] : 0.0f;
#pragma unroll
  for (int g = 0; g < 4; ++g) {
    const int col = tn + g * 16 + l;
    const float bval = bias ? bias[col] : 0.0f;
#pragma unroll
    for (int r = 0; r < 8; ++r) {
      float v = acc[g][r] + bval;
      if (act == 1)      v = v > 0.0f ? v : 0.0f;
      else if (act == 2) v = v >= 0.0f ? v : slope * v;
      size_t off = (size_t)(m0 + r) * N + col;
      if (D)   D[off] = v;
      if (Dbf) Dbf[off] = f2bf(v);
    }
  }
}

// ---------------------------------------------------------------------------
// Small helper kernels
// ---------------------------------------------------------------------------
__global__ void cvt_f32_bf16(const float* __restrict__ in, bf16_t* __restrict__ out,
                             size_t n) {
  size_t i = (size_t)blockIdx.x * blockDim.x + threadIdx.x;
  if (i < n) out[i] = f2bf(in[i]);
}

// per-node attention logits a_src/a_dst: h[N,1024] dot att[4,256] per head
__global__ __launch_bounds__(1024)
void node_att(const float* __restrict__ h, const float* __restrict__ att_src,
              const float* __restrict__ att_dst, float* __restrict__ a_src,
              float* __restrict__ a_dst) {
  __shared__ float s[1024];
  const int node = blockIdx.x, t = threadIdx.x;
  const float v = h[(size_t)node * 1024 + t];
  s[t] = v * att_src[t];
  __syncthreads();
  for (int off = 128; off > 0; off >>= 1) {
    if ((t & 255) < off) s[t] += s[t + off];
    __syncthreads();
  }
  if ((t & 255) == 0) a_src[node * 4 + (t >> 8)] = s[t];
  __syncthreads();
  s[t] = v * att_dst[t];
  __syncthreads();
  for (int off = 128; off > 0; off >>= 1) {
    if ((t & 255) < off) s[t] += s[t + off];
    __syncthreads();
  }
  if ((t & 255) == 0) a_dst[node * 4 + (t >> 8)] = s[t];
}

// c[h] = sum_c Wed[h*256+c] * ae[h*256+c]   (lin_edge on scalar is rank-1)
__global__ __launch_bounds__(1024)
void edge_coef(const float* __restrict__ Wed, const float* __restrict__ ae,
               float* __restrict__ coef) {
  __shared__ float s[1024];
  const int t = threadIdx.x;
  s[t] = Wed[t] * ae[t];
  __syncthreads();
  for (int off = 128; off > 0; off >>= 1) {
    if ((t & 255) < off) s[t] += s[t + off];
    __syncthreads();
  }
  if ((t & 255) == 0) coef[t >> 8] = s[t];
}

// edge gate: sigmoid(prelu_emb @ Wp + bp)
__global__ void edge_gate(const float* __restrict__ ee, const float* __restrict__ Wp,
                          const float* __restrict__ bp, float* __restrict__ ew, int E) {
  int e = blockIdx.x * blockDim.x + threadIdx.x;
  if (e >= E) return;
  float s = bp[0];
#pragma unroll 8
  for (int i = 0; i < 64; ++i) s += ee[(size_t)e * 64 + i] * Wp[i];
  ew[e] = 1.0f / (1.0f + __expf(-s));
}

__global__ void reduce_sum(const float* __restrict__ x, int n, float* __restrict__ out) {
  __shared__ float s[256];
  float acc = 0.0f;
  for (int i = blockIdx.x * 256 + threadIdx.x; i < n; i += gridDim.x * 256) acc += x[i];
  s[threadIdx.x] = acc;
  __syncthreads();
  for (int off = 128; off > 0; off >>= 1) {
    if (threadIdx.x < off) s[threadIdx.x] += s[threadIdx.x + off];
    __syncthreads();
  }
  if (threadIdx.x == 0) atomicAdd(out, s[0]);
}

__global__ void fill_mean(float* __restrict__ tail, const float* __restrict__ sum,
                          float inv, int n) {
  int i = blockIdx.x * blockDim.x + threadIdx.x;
  if (i < n) tail[i] = sum[0] * inv;
}

// pass 1: alpha = leakyrelu(a_src[s]+a_dst[d]+ew*c[h]); segment max via uint keys
__global__ void alpha_max(const int* __restrict__ ei, const float* __restrict__ ew,
                          const float* __restrict__ a_src, const float* __restrict__ a_dst,
                          const float* __restrict__ coef, float* __restrict__ alpha,
                          unsigned* __restrict__ segmax, int Etot, int Ereal) {
  int idx = blockIdx.x * blockDim.x + threadIdx.x;
  if (idx >= Etot * 4) return;
  int e = idx >> 2, h = idx & 3;
  int sN, dN;
  if (e < Ereal) { sN = ei[e]; dN = ei[Ereal + e]; } else { sN = dN = e - Ereal; }
  float a = a_src[sN * 4 + h] + a_dst[dN * 4 + h] + ew[e] * coef[h];
  a = a >= 0.0f ? a : NEG_SLOPE * a;
  alpha[idx] = a;
  atomicMax(&segmax[dN * 4 + h], fkey(a));
}

// pass 2: alpha <- exp(alpha - max); segment sum
__global__ void alpha_exp(const int* __restrict__ ei, float* __restrict__ alpha,
                          const unsigned* __restrict__ segmax, float* __restrict__ segsum,
                          int Etot, int Ereal) {
  int idx = blockIdx.x * blockDim.x + threadIdx.x;
  if (idx >= Etot * 4) return;
  int e = idx >> 2, h = idx & 3;
  int dN = (e < Ereal) ? ei[Ereal + e] : (e - Ereal);
  float ev = __expf(alpha[idx] - keyf(segmax[dN * 4 + h]));
  alpha[idx] = ev;
  atomicAdd(&segsum[dN * 4 + h], ev);
}

// pass 3: out[dst, 4ch] += h[src, 4ch] * alpha/(segsum+eps)
// one thread per (edge, 4-channel group): float4 gather + 4 atomic adds
__global__ void aggregate(const int* __restrict__ ei, const float* __restrict__ h,
                          const float* __restrict__ alpha, const float* __restrict__ segsum,
                          float* __restrict__ out, int Etot, int Ereal) {
  int idx = blockIdx.x * blockDim.x + threadIdx.x;
  if (idx >= Etot * 256) return;
  int e = idx >> 8, q = idx & 255;      // q = 4-channel group in [0,256)
  int hh = q >> 6;                       // head = (q*4)/256
  int sN, dN;
  if (e < Ereal) { sN = ei[e]; dN = ei[Ereal + e]; } else { sN = dN = e - Ereal; }
  float coeff = alpha[e * 4 + hh] / (segsum[dN * 4 + hh] + 1e-16f);
  const float4 hv = *(const float4*)(h + (size_t)sN * 1024 + q * 4);
  float* op = out + (size_t)dN * 1024 + q * 4;
  atomicAdd(op + 0, hv.x * coeff);
  atomicAdd(op + 1, hv.y * coeff);
  atomicAdd(op + 2, hv.z * coeff);
  atomicAdd(op + 3, hv.w * coeff);
}

// out = prelu(in + bias); write f32 (in-place ok) and bf16
__global__ void bias_prelu_cvt(const float* __restrict__ in, const float* __restrict__ bias,
                               const float* __restrict__ pa, float* __restrict__ outf,
                               bf16_t* __restrict__ outb, size_t n, int dim) {
  size_t i = (size_t)blockIdx.x * blockDim.x + threadIdx.x;
  if (i >= n) return;
  float v = in[i] + bias[i % dim];
  float a = pa[0];
  v = v >= 0.0f ? v : a * v;
  outf[i] = v;
  outb[i] = f2bf(v);
}

// y = prelu(LN(out1 + b1 + x1)) -> bf16
__global__ __launch_bounds__(1024)
void resid_ln_prelu(const float* __restrict__ out1, const float* __restrict__ b1,
                    const float* __restrict__ x1, const float* __restrict__ g,
                    const float* __restrict__ bta, const float* __restrict__ pa,
                    bf16_t* __restrict__ zb) {
  __shared__ float s[1024];
  __shared__ float mu_s, rs_s;
  const int node = blockIdx.x, t = threadIdx.x;
  const size_t base = (size_t)node * 1024;
  float v = out1[base + t] + b1[t] + x1[base + t];
  s[t] = v;
  __syncthreads();
  for (int off = 512; off > 0; off >>= 1) {
    if (t < off) s[t] += s[t + off];
    __syncthreads();
  }
  if (t == 0) mu_s = s[0] * (1.0f / 1024.0f);
  __syncthreads();
  float d = v - mu_s;
  s[t] = d * d;
  __syncthreads();
  for (int off = 512; off > 0; off >>= 1) {
    if (t < off) s[t] += s[t + off];
    __syncthreads();
  }
  if (t == 0) rs_s = rsqrtf(s[0] * (1.0f / 1024.0f) + 1e-5f);
  __syncthreads();
  float r = d * rs_s * g[t] + bta[t];
  float a = pa[0];
  r = r >= 0.0f ? r : a * r;
  zb[base + t] = f2bf(r);
}

// out[n] = dot(o1[n,:256], Wo2) + bo2
__global__ __launch_bounds__(256)
void final_dot(const float* __restrict__ o1, const float* __restrict__ Wo2,
               const float* __restrict__ bo2, float* __restrict__ out) {
  __shared__ float s[256];
  const int node = blockIdx.x, t = threadIdx.x;
  s[t] = o1[(size_t)node * 256 + t] * Wo2[t];
  __syncthreads();
  for (int off = 128; off > 0; off >>= 1) {
    if (t < off) s[t] += s[t + off];
    __syncthreads();
  }
  if (t == 0) out[node] = s[0] + bo2[0];
}

// ---------------------------------------------------------------------------
// Host-side orchestration
// ---------------------------------------------------------------------------
static inline int cdiv(long long a, long long b) { return (int)((a + b - 1) / b); }

extern "C" void kernel_launch(void* const* d_in, const int* in_sizes, int n_in,
                              void* d_out, int out_size, void* d_ws, size_t ws_size,
                              hipStream_t stream) {
  const int N  = in_sizes[0] / 512;   // 20000
  const int E  = in_sizes[1] / 2;     // 120000
  const int ET = E + N;               // edges incl. self loops

  const float* x    = (const float*)d_in[0];
  const int*   ei   = (const int*)d_in[1];
  const float* ea   = (const float*)d_in[2];
  const float* Wn   = (const float*)d_in[3];
  const float* bn   = (const float*)d_in[4];
  const float* We   = (const float*)d_in[5];
  const float* be   = (const float*)d_in[6];
  const float* Wp   = (const float*)d_in[7];
  const float* bp   = (const float*)d_in[8];
  const float* pa   = (const float*)d_in[9];
  const float* W0   = (const float*)d_in[10];
  const float* as0  = (const float*)d_in[11];
  const float* ad0  = (const float*)d_in[12];
  const float* Wed0 = (const float*)d_in[13];
  const float* ae0  = (const float*)d_in[14];
  const float* b0   = (const float*)d_in[15];
  const float* W1   = (const float*)d_in[16];
  const float* as1  = (const float*)d_in[17];
  const float* ad1  = (const float*)d_in[18];
  const float* Wed1 = (const float*)d_in[19];
  const float* ae1  = (const float*)d_in[20];
  const float* b1   = (const float*)d_in[21];
  const float* lng  = (const float*)d_in[22];
  const float* lnb  = (const float*)d_in[23];
  const float* Wo1  = (const float*)d_in[24];
  const float* bo1  = (const float*)d_in[25];
  const float* Wo2  = (const float*)d_in[26];
  const float* bo2  = (const float*)d_in[27];
  float* out = (float*)d_out;

  // ---- workspace layout -----------------------------------------------
  char* p = (char*)d_ws;
  auto alloc = [&](size_t bytes) -> void* {
    void* r = (void*)p;
    p += (bytes + 255) & ~(size_t)255;
    return r;
  };
  bf16_t* xb    = (bf16_t*)alloc((size_t)N * 512 * 2);
  bf16_t* Wnp   = (bf16_t*)alloc((size_t)512 * 256 * 2);    // packed B fragments
  bf16_t* xhb   = (bf16_t*)alloc((size_t)N * 256 * 2);
  bf16_t* eab   = (bf16_t*)alloc((size_t)E * 64 * 2);
  bf16_t* Wep   = (bf16_t*)alloc((size_t)64 * 64 * 2);
  float*  ee    = (float*)alloc((size_t)E * 64 * 4);        // reused later for o1 [N,256]
  float*  ew    = (float*)alloc((size_t)ET * 4);
  float*  esum  = (float*)alloc(256);
  bf16_t* W0p   = (bf16_t*)alloc((size_t)256 * 1024 * 2);
  bf16_t* W1p   = (bf16_t*)alloc((size_t)1024 * 1024 * 2);
  bf16_t* Wo1p  = (bf16_t*)alloc((size_t)1024 * 256 * 2);
  float*  bufA  = (float*)alloc((size_t)N * 1024 * 4);      // h0, then h1
  float*  bufB  = (float*)alloc((size_t)N * 1024 * 4);      // out0 -> x1 (in place)
  float*  bufC  = (float*)alloc((size_t)N * 1024 * 4);      // out1
  bf16_t* x1b   = (bf16_t*)alloc((size_t)N * 1024 * 2);     // x1 bf16, reused as zb
  float*  asrc  = (float*)alloc((size_t)N * 4 * 4);
  float*  adst  = (float*)alloc((size_t)N * 4 * 4);
  float*  coef  = (float*)alloc(256);
  float*  alpha = (float*)alloc((size_t)ET * 4 * 4);
  unsigned* segmax = (unsigned*)alloc((size_t)N * 4 * 4);
  float*  segsum = (float*)alloc((size_t)N * 4 * 4);
  float*  o1    = ee;                                       // reuse edge_emb buffer

  // ---- stage 0: conversions / weight fragment packing -----------------
  cvt_f32_bf16<<<cdiv((size_t)N * 512, 256), 256, 0, stream>>>(x, xb, (size_t)N * 512);
  cvt_f32_bf16<<<cdiv((size_t)E * 64, 256), 256, 0, stream>>>(ea, eab, (size_t)E * 64);
  pack_b<<<cdiv(256 * 512, 256), 256, 0, stream>>>(Wn, Wnp, 256, 512);
  pack_b<<<cdiv(64 * 64, 256), 256, 0, stream>>>(We, Wep, 64, 64);
  pack_b<<<cdiv(1024 * 256, 256), 256, 0, stream>>>(W0, W0p, 1024, 256);
  pack_b<<<cdiv(1024 * 1024, 256), 256, 0, stream>>>(W1, W1p, 1024, 1024);
  pack_b<<<cdiv(256 * 1024, 256), 256, 0, stream>>>(Wo1, Wo1p, 256, 1024);

  // ---- node_lin + relu -> xhb (bf16) ----------------------------------
  wmma_gemm_bf16<<<cdiv((N / 16) * (256 / 64), 8), 256, 0, stream>>>(
      xb, Wnp, bn, pa, nullptr, xhb, N, 256, 512, /*act=*/1);

  // ---- edge MLP: prelu(edge_attr @ We.T + be) -> ee (f32) -------------
  wmma_gemm_bf16<<<cdiv((E / 16) * (64 / 64), 8), 256, 0, stream>>>(
      eab, Wep, be, pa, ee, nullptr, E, 64, 64, /*act=*/2);

  // ---- edge gate + mean fill for self loops ---------------------------
  edge_gate<<<cdiv(E, 256), 256, 0, stream>>>(ee, Wp, bp, ew, E);
  hipMemsetAsync(esum, 0, sizeof(float), stream);
  reduce_sum<<<256, 256, 0, stream>>>(ew, E, esum);
  fill_mean<<<cdiv(N, 256), 256, 0, stream>>>(ew + E, esum, 1.0f / (float)E, N);

  // =====================  GAT layer 0  =================================
  wmma_gemm_bf16<<<cdiv((N / 16) * (1024 / 64), 8), 256, 0, stream>>>(
      xhb, W0p, nullptr, pa, bufA, nullptr, N, 1024, 256, /*act=*/0);      // h0
  node_att<<<N, 1024, 0, stream>>>(bufA, as0, ad0, asrc, adst);
  edge_coef<<<1, 1024, 0, stream>>>(Wed0, ae0, coef);
  hipMemsetAsync(segmax, 0, (size_t)N * 4 * 4, stream);
  hipMemsetAsync(segsum, 0, (size_t)N * 4 * 4, stream);
  alpha_max<<<cdiv((size_t)ET * 4, 256), 256, 0, stream>>>(ei, ew, asrc, adst, coef,
                                                           alpha, segmax, ET, E);
  alpha_exp<<<cdiv((size_t)ET * 4, 256), 256, 0, stream>>>(ei, alpha, segmax, segsum, ET, E);
  hipMemsetAsync(bufB, 0, (size_t)N * 1024 * 4, stream);
  aggregate<<<cdiv((size_t)ET * 256, 256), 256, 0, stream>>>(ei, bufA, alpha, segsum,
                                                             bufB, ET, E);
  // x1 = prelu(out0 + b0)  (in place) + bf16 copy
  bias_prelu_cvt<<<cdiv((size_t)N * 1024, 256), 256, 0, stream>>>(
      bufB, b0, pa, bufB, x1b, (size_t)N * 1024, 1024);

  // =====================  GAT layer 1  =================================
  wmma_gemm_bf16<<<cdiv((N / 16) * (1024 / 64), 8), 256, 0, stream>>>(
      x1b, W1p, nullptr, pa, bufA, nullptr, N, 1024, 1024, /*act=*/0);     // h1
  node_att<<<N, 1024, 0, stream>>>(bufA, as1, ad1, asrc, adst);
  edge_coef<<<1, 1024, 0, stream>>>(Wed1, ae1, coef);
  hipMemsetAsync(segmax, 0, (size_t)N * 4 * 4, stream);
  hipMemsetAsync(segsum, 0, (size_t)N * 4 * 4, stream);
  alpha_max<<<cdiv((size_t)ET * 4, 256), 256, 0, stream>>>(ei, ew, asrc, adst, coef,
                                                           alpha, segmax, ET, E);
  alpha_exp<<<cdiv((size_t)ET * 4, 256), 256, 0, stream>>>(ei, alpha, segmax, segsum, ET, E);
  hipMemsetAsync(bufC, 0, (size_t)N * 1024 * 4, stream);
  aggregate<<<cdiv((size_t)ET * 256, 256), 256, 0, stream>>>(ei, bufA, alpha, segsum,
                                                             bufC, ET, E);

  // residual + LN + prelu -> zb (reuses x1b storage)
  resid_ln_prelu<<<N, 1024, 0, stream>>>(bufC, b1, bufB, lng, lnb, pa, x1b);

  // head: prelu(z @ Wo1.T + bo1) -> o1 f32, then dot with Wo2
  wmma_gemm_bf16<<<cdiv((N / 16) * (256 / 64), 8), 256, 0, stream>>>(
      x1b, Wo1p, bo1, pa, o1, nullptr, N, 256, 1024, /*act=*/2);
  final_dot<<<N, 256, 0, stream>>>(o1, Wo2, bo2, out);
}
